// SynthesisBlock_14474039788060
// MI455X (gfx1250) — compile-verified, matrix operand order checked by reading, not run on hardware
//
#include <hip/hip_runtime.h>
#include <hip/hip_bf16.h>

typedef __attribute__((ext_vector_type(16))) _Float16 v16h;
typedef __attribute__((ext_vector_type(8)))  _Float16 v8h;
typedef __attribute__((ext_vector_type(8)))  float    v8f;

union H16 { v16h v; v8h h2[2]; _Float16 e[16]; };
union F8  { v8f  v; float e[8]; };

#define CIN   512
#define COUT  512
#define BATCH 8
#define RES   64
#define KTOT  4608   // 9 * 512
#define SG    0.04419417382415922f  // 1/sqrt(512)

// ---------------------------------------------------------------------------
// CDNA5 async global->LDS direct copy (16B per active lane), ASYNCcnt-tracked.
// ---------------------------------------------------------------------------
__device__ __forceinline__ void async_load_b128(unsigned lds_off, const void* gaddr) {
  asm volatile("global_load_async_to_lds_b128 %0, %1, off"
               :: "v"(lds_off), "v"((unsigned long long)(size_t)gaddr)
               : "memory");
}
__device__ __forceinline__ void wait_async0() {
  asm volatile("s_wait_asynccnt 0x0" ::: "memory");
}

// ---------------------------------------------------------------------------
// styles: s0/s1/srgb [B,512]  (srgb carries the extra 1/sqrt(Cin) factor)
// ---------------------------------------------------------------------------
__global__ void styles_kernel(const float* __restrict__ ws,
                              const float* __restrict__ a0w, const float* __restrict__ a0b,
                              const float* __restrict__ a1w, const float* __restrict__ a1b,
                              const float* __restrict__ agw, const float* __restrict__ agb,
                              float* __restrict__ s0, float* __restrict__ s1,
                              float* __restrict__ srgb) {
  int t = blockIdx.x * 256 + threadIdx.x;
  if (t >= 3 * BATCH * 512) return;
  int type = t / (BATCH * 512);
  int rem  = t - type * (BATCH * 512);
  int b = rem >> 9, c = rem & 511;
  const float* aw = (type == 0) ? a0w : (type == 1) ? a1w : agw;
  const float* ab = (type == 0) ? a0b : (type == 1) ? a1b : agb;
  const float* wv = ws + (b * 3 + type) * 512;
  const float* ar = aw + (size_t)c * 512;
  float acc = 0.f;
  for (int j = 0; j < 512; ++j) acc = fmaf(wv[j], ar[j], acc);
  float s = acc * SG + ab[c];
  if (type == 2) s *= SG;
  float* dst = (type == 0) ? s0 : (type == 1) ? s1 : srgb;
  dst[b * 512 + c] = s;
}

// ---------------------------------------------------------------------------
// weight repack -> f16, K-major k = (ky*3+kx)*512 + ci. conv0 spatially flipped.
// ---------------------------------------------------------------------------
__global__ void wprep_kernel(const float* __restrict__ w0, const float* __restrict__ w1,
                             _Float16* __restrict__ wf0, _Float16* __restrict__ wf1) {
  int t = blockIdx.x * 256 + threadIdx.x;
  if (t >= COUT * KTOT) return;
  int co = t / KTOT, k = t - co * KTOT;
  int s = k / 512, ci = k - s * 512;
  int ky = s / 3, kx = s - ky * 3;
  wf0[t] = (_Float16)w0[(((size_t)co * 512 + ci) * 3 + (2 - ky)) * 3 + (2 - kx)];
  wf1[t] = (_Float16)w1[(((size_t)co * 512 + ci) * 3 + ky) * 3 + kx];
}

// ---------------------------------------------------------------------------
// demod: dcoef[b,co] = rsqrt( sum_ci s[b,ci]^2 * sum_k w[co,ci,k]^2 + 1e-8 )
// ---------------------------------------------------------------------------
__global__ void dcoef_kernel(const float* __restrict__ w, const float* __restrict__ s,
                             float* __restrict__ dc) {
  int t = blockIdx.x * 256 + threadIdx.x;
  if (t >= BATCH * COUT) return;
  int b = t >> 9, co = t & 511;
  const float* sb = s + b * 512;
  float acc = 0.f;
  for (int ci = 0; ci < 512; ++ci) {
    const float* wp = w + ((size_t)co * 512 + ci) * 9;
    float w2 = 0.f;
    for (int j = 0; j < 9; ++j) w2 = fmaf(wp[j], wp[j], w2);
    acc = fmaf(sb[ci] * sb[ci], w2, acc);
  }
  dc[t] = rsqrtf(acc + 1e-8f);
}

// ---------------------------------------------------------------------------
// modulate + zero-insert upsample2 + FIR [1,3,3,1]/8 (gain 4), pad (3,2):
// xu[b][66][66][512] f16 (channel-last so GEMM B tiles are contiguous)
// ---------------------------------------------------------------------------
__global__ void upx_kernel(const float* __restrict__ x, const float* __restrict__ s0,
                           _Float16* __restrict__ xu) {
  long t = (long)blockIdx.x * 256 + threadIdx.x;
  const long total = (long)BATCH * 66 * 66 * 512;
  if (t >= total) return;
  int ci = (int)(t & 511);
  long pix = t >> 9;
  int xx = (int)(pix % 66); pix /= 66;
  int yy = (int)(pix % 66);
  int b  = (int)(pix / 66);
  const float F[4] = {0.125f, 0.375f, 0.375f, 0.125f};
  float acc = 0.f;
  for (int iy = 0; iy < 4; ++iy) {
    int p = yy + iy;
    if (!(p & 1)) continue;              // zero-inserted sample present only at odd p
    int h = (p - 3) >> 1;
    if ((unsigned)h >= 32u) continue;
    float ra = 0.f;
    for (int ix = 0; ix < 4; ++ix) {
      int q = xx + ix;
      if (!(q & 1)) continue;
      int wi = (q - 3) >> 1;
      if ((unsigned)wi >= 32u) continue;
      ra = fmaf(F[ix], x[(((size_t)b * 512 + ci) * 32 + h) * 32 + wi], ra);
    }
    acc = fmaf(F[iy], ra, acc);
  }
  xu[t] = (_Float16)(4.f * acc * s0[b * 512 + ci]);
}

// ---------------------------------------------------------------------------
// GEMM helpers: async fill of one K-chunk (32 halves) into an LDS buffer, and
// 4-WMMA compute of one K-chunk from an LDS buffer.
// ---------------------------------------------------------------------------
template <bool PAD>
__device__ __forceinline__ void issue_tile(int kc, _Float16 (*As)[32], _Float16 (*Bs)[32],
                                           const _Float16* __restrict__ Wf,
                                           const _Float16* __restrict__ Xin,
                                           int cob, int b, int y, int fr, int fc) {
  const int IW  = PAD ? 64 : 66;
  const int s   = kc >> 4;
  const int ci0 = (kc & 15) * 32;
  const int ky = s / 3, kx = s - 3 * (s / 3);
  async_load_b128((unsigned)(size_t)&As[fr][fc],
                  &Wf[(size_t)(cob + fr) * KTOT + (size_t)kc * 32 + fc]);
  async_load_b128((unsigned)(size_t)&As[fr + 64][fc],
                  &Wf[(size_t)(cob + fr + 64) * KTOT + (size_t)kc * 32 + fc]);
  const int yy = y + ky - (PAD ? 1 : 0);
  const int xx = fr + kx - (PAD ? 1 : 0);
  if (!PAD || ((unsigned)yy < 64u && (unsigned)xx < 64u)) {
    async_load_b128((unsigned)(size_t)&Bs[fr][fc],
                    &Xin[(((size_t)(b * IW + yy)) * IW + xx) * 512 + ci0 + fc]);
  } else {
    v8h z = {};
    *(v8h*)&Bs[fr][fc] = z;              // zero-fill pad region (DScnt path)
  }
}

__device__ __forceinline__ void compute_tile(const _Float16 (*As)[32], const _Float16 (*Bs)[32],
                                             int mo, int no, int ml, int kb,
                                             v8f& a00, v8f& a10, v8f& a01, v8f& a11) {
  // Fragments per ISA 7.12.2: lane<16 K={0..7,16..23}, lane>=16 K={8..15,24..31}
  H16 fa0, fa1, fb0, fb1;
  fa0.h2[0] = *(const v8h*)&As[mo + ml][kb];
  fa0.h2[1] = *(const v8h*)&As[mo + ml][kb + 16];
  fa1.h2[0] = *(const v8h*)&As[mo + 16 + ml][kb];
  fa1.h2[1] = *(const v8h*)&As[mo + 16 + ml][kb + 16];
  fb0.h2[0] = *(const v8h*)&Bs[no + ml][kb];
  fb0.h2[1] = *(const v8h*)&Bs[no + ml][kb + 16];
  fb1.h2[0] = *(const v8h*)&Bs[no + 16 + ml][kb];
  fb1.h2[1] = *(const v8h*)&Bs[no + 16 + ml][kb + 16];
  a00 = __builtin_amdgcn_wmma_f32_16x16x32_f16(false, fa0.v, false, fb0.v,
                                               (short)0, a00, false, false);
  a10 = __builtin_amdgcn_wmma_f32_16x16x32_f16(false, fa1.v, false, fb0.v,
                                               (short)0, a10, false, false);
  a01 = __builtin_amdgcn_wmma_f32_16x16x32_f16(false, fa0.v, false, fb1.v,
                                               (short)0, a01, false, false);
  a11 = __builtin_amdgcn_wmma_f32_16x16x32_f16(false, fa1.v, false, fb1.v,
                                               (short)0, a11, false, false);
}

// ---------------------------------------------------------------------------
// Implicit-GEMM modulated conv via V_WMMA_F32_16X16X32_F16.
// M = Cout (A = weights), N = pixels of one image row, K = 9*512.
// Block: 256 thr = 8 waves, tile 128(co) x 64(x) for one (b,y).
// Per wave: 2Mx2N = 4 WMMA tiles. Double-buffered LDS with async DMA:
//   wait_asynccnt 0 -> barrier -> issue k+1 (other buffer) -> compute k.
// One barrier per K-step; DMA for k+1 overlaps WMMA for k.
// MODE 0: conv0 -> f16 a0m[b][y][x][co] pre-scaled by s1.
// MODE 1: conv1 -> f32 d_out x[b][co][y][x].
// ---------------------------------------------------------------------------
template <bool PAD, int MODE>
__global__ __launch_bounds__(256) void gemm_conv(
    const _Float16* __restrict__ Wf,    // [512][4608]
    const _Float16* __restrict__ Xin,   // MODE0: xu [8][66][66][512]; MODE1: a0m [8][64][64][512]
    const float* __restrict__ dcoef,    // [8][512]
    const float* __restrict__ bias,     // [512]
    const float* __restrict__ nstr,     // [1]
    const float* __restrict__ nconst,   // [64][64]
    const float* __restrict__ smul,     // s1 (MODE0 only)
    _Float16* __restrict__ outH,        // MODE0
    float* __restrict__ outF) {         // MODE1
  __shared__ __align__(16) _Float16 As0[128][32];
  __shared__ __align__(16) _Float16 As1[128][32];
  __shared__ __align__(16) _Float16 Bs0[64][32];
  __shared__ __align__(16) _Float16 Bs1[64][32];

  const int t = threadIdx.x;
  const int lane = t & 31;
  const int wave = t >> 5;
  const int ml = lane & 15;
  const int kb = (lane >> 4) * 8;
  const int mo = (wave & 3) * 32;          // wave's M offset (2 sub-tiles)
  const int no = (wave >> 2) * 32;         // wave's N offset (2 sub-tiles)
  const int cob = blockIdx.x * 128;        // Cout block base
  const int b = blockIdx.y >> 6;
  const int y = blockIdx.y & 63;
  const int fr = t >> 2;                   // fill row (0..63)
  const int fc = (t & 3) * 8;              // fill col (0,8,16,24)

  v8f a00 = {}, a10 = {}, a01 = {}, a11 = {};

  // Prologue: start DMA for chunk 0 into buffer 0.
  issue_tile<PAD>(0, As0, Bs0, Wf, Xin, cob, b, y, fr, fc);

  // 144 chunks, even count -> manual ping-pong with static buffers.
#pragma unroll 1
  for (int kc = 0; kc < 144; kc += 2) {
    wait_async0();
    __syncthreads();                       // buf0 ready for all waves; buf1 free
    issue_tile<PAD>(kc + 1, As1, Bs1, Wf, Xin, cob, b, y, fr, fc);
    compute_tile(As0, Bs0, mo, no, ml, kb, a00, a10, a01, a11);

    wait_async0();
    __syncthreads();                       // buf1 ready; buf0 free
    if (kc + 2 < 144)
      issue_tile<PAD>(kc + 2, As0, Bs0, Wf, Xin, cob, b, y, fr, fc);
    compute_tile(As1, Bs1, mo, no, ml, kb, a00, a10, a01, a11);
  }

  // Epilogue: demod, noise, bias, lrelu*sqrt2, clamp, (MODE0: * s1 -> f16 NHWC)
  const float ns = nstr[0];
  F8 A[4];
  A[0].v = a00; A[1].v = a10; A[2].v = a01; A[3].v = a11;
#pragma unroll
  for (int ti = 0; ti < 4; ++ti) {
    const int mofs = mo + (ti & 1) * 16;
    const int nofs = no + (ti >> 1) * 16;
    const int xp = nofs + ml;
    const float nz = nconst[y * 64 + xp] * ns;
#pragma unroll
    for (int r = 0; r < 8; ++r) {
      const int co = cob + mofs + r + ((lane >> 4) << 3);  // D layout: VGPR r -> M=r / 8+r
      float v = A[ti].e[r];
      v = v * dcoef[b * 512 + co] + nz + bias[co];
      v = (v >= 0.f ? v : 0.2f * v) * 1.4142135623730951f;
      v = fminf(fmaxf(v, -256.f), 256.f);
      if (MODE == 0) {
        outH[(((size_t)(b * 64 + y) * 64 + xp) * 512) + co] =
            (_Float16)(v * smul[b * 512 + co]);
      } else {
        outF[(((size_t)(b * 512 + co) * 64 + y) * 64) + xp] = v;
      }
    }
  }
}

// ---------------------------------------------------------------------------
// toRGB (1x1 mod conv, no demod) + FIR-upsampled skip image, fused.
// grid = B*64 rows, 64 threads (x) -> coalesced reads of act over x.
// ---------------------------------------------------------------------------
__global__ void rgb_kernel(const float* __restrict__ actx, const float* __restrict__ srgb,
                           const float* __restrict__ wrgb, const float* __restrict__ brgb,
                           const float* __restrict__ img, float* __restrict__ imgout) {
  const int b = blockIdx.x >> 6, y = blockIdx.x & 63;
  const int x = threadIdx.x;
  float a0 = 0.f, a1 = 0.f, a2 = 0.f;
  for (int co = 0; co < 512; ++co) {
    float a = actx[(((size_t)b * 512 + co) * 64 + y) * 64 + x] * srgb[b * 512 + co];
    a0 = fmaf(wrgb[co], a, a0);
    a1 = fmaf(wrgb[512 + co], a, a1);
    a2 = fmaf(wrgb[1024 + co], a, a2);
  }
  const float F[4] = {0.125f, 0.375f, 0.375f, 0.125f};
  for (int ch = 0; ch < 3; ++ch) {
    float v = (ch == 0 ? a0 : ch == 1 ? a1 : a2) + brgb[ch];
    v = fminf(fmaxf(v, -256.f), 256.f);
    // upsample2d(img): pad (2,1), gain 4 -> nonzero taps at even padded index
    float up = 0.f;
    for (int iy = 0; iy < 4; ++iy) {
      int p = y + iy;
      if (p & 1) continue;
      int h = (p - 2) >> 1;
      if ((unsigned)h >= 32u) continue;
      float ra = 0.f;
      for (int ix = 0; ix < 4; ++ix) {
        int q = x + ix;
        if (q & 1) continue;
        int w = (q - 2) >> 1;
        if ((unsigned)w >= 32u) continue;
        ra = fmaf(F[ix], img[(((size_t)b * 3 + ch) * 32 + h) * 32 + w], ra);
      }
      up = fmaf(F[iy], ra, up);
    }
    imgout[(((size_t)b * 3 + ch) * 64 + y) * 64 + x] = 4.f * up + v;
  }
}

// ---------------------------------------------------------------------------
extern "C" void kernel_launch(void* const* d_in, const int* in_sizes, int n_in,
                              void* d_out, int out_size, void* d_ws, size_t ws_size,
                              hipStream_t stream) {
  const float* x    = (const float*)d_in[0];
  const float* img  = (const float*)d_in[1];
  const float* wsv  = (const float*)d_in[2];
  const float* a0w  = (const float*)d_in[3];
  const float* a0b  = (const float*)d_in[4];
  const float* w0   = (const float*)d_in[5];
  const float* b0   = (const float*)d_in[6];
  const float* ns0  = (const float*)d_in[7];
  const float* nc0  = (const float*)d_in[8];
  const float* a1w  = (const float*)d_in[9];
  const float* a1b  = (const float*)d_in[10];
  const float* w1   = (const float*)d_in[11];
  const float* b1   = (const float*)d_in[12];
  const float* ns1  = (const float*)d_in[13];
  const float* nc1  = (const float*)d_in[14];
  const float* agw  = (const float*)d_in[15];
  const float* agb  = (const float*)d_in[16];
  const float* wrgb = (const float*)d_in[17];
  const float* brgb = (const float*)d_in[18];

  float* out_x   = (float*)d_out;                              // [8,512,64,64]
  float* out_img = out_x + (size_t)BATCH * COUT * RES * RES;   // [8,3,64,64]

  // carve workspace (256B aligned slices)
  char* base = (char*)d_ws;
  size_t off = 0;
  auto carve = [&](size_t bytes) {
    void* p = base + off;
    off = (off + bytes + 255) & ~(size_t)255;
    return p;
  };
  float*    s0     = (float*)carve((size_t)BATCH * 512 * 4);
  float*    s1     = (float*)carve((size_t)BATCH * 512 * 4);
  float*    srgb   = (float*)carve((size_t)BATCH * 512 * 4);
  float*    dc0    = (float*)carve((size_t)BATCH * 512 * 4);
  float*    dc1    = (float*)carve((size_t)BATCH * 512 * 4);
  _Float16* wf0    = (_Float16*)carve((size_t)COUT * KTOT * 2);
  _Float16* wf1    = (_Float16*)carve((size_t)COUT * KTOT * 2);
  _Float16* xu     = (_Float16*)carve((size_t)BATCH * 66 * 66 * 512 * 2);
  _Float16* a0m    = (_Float16*)carve((size_t)BATCH * 64 * 64 * 512 * 2);

  // 1) styles
  styles_kernel<<<(3 * BATCH * 512 + 255) / 256, 256, 0, stream>>>(
      wsv, a0w, a0b, a1w, a1b, agw, agb, s0, s1, srgb);
  // 2) weight repack (f16, K-major, conv0 flipped)
  wprep_kernel<<<(COUT * KTOT + 255) / 256, 256, 0, stream>>>(w0, w1, wf0, wf1);
  // 3) demod coefficients
  dcoef_kernel<<<(BATCH * COUT + 255) / 256, 256, 0, stream>>>(w0, s0, dc0);
  dcoef_kernel<<<(BATCH * COUT + 255) / 256, 256, 0, stream>>>(w1, s1, dc1);
  // 4) modulate + upfirdn up2 -> xu (f16 NHWC)
  {
    long total = (long)BATCH * 66 * 66 * 512;
    upx_kernel<<<(int)((total + 255) / 256), 256, 0, stream>>>(x, s0, xu);
  }
  // 5) conv0 (WMMA) -> a0m (f16 NHWC, pre-modulated by s1)
  dim3 gg(COUT / 128, BATCH * RES);
  gemm_conv<false, 0><<<gg, 256, 0, stream>>>(wf0, xu, dc0, b0, ns0, nc0, s1,
                                              a0m, nullptr);
  // 6) conv1 (WMMA, pad=1) -> d_out x (f32 NCHW)
  gemm_conv<true, 1><<<gg, 256, 0, stream>>>(wf1, a0m, dc1, b1, ns1, nc1, nullptr,
                                             nullptr, out_x);
  // 7) toRGB + upsampled skip image -> d_out img
  rgb_kernel<<<BATCH * RES, 64, 0, stream>>>(out_x, srgb, wrgb, brgb, img, out_img);
}
